// AnalysisTransform_81484119539717
// MI455X (gfx1250) — compile-verified, matrix operand order checked by reading, not run on hardware
//
#include <hip/hip_runtime.h>

// ---------------------------------------------------------------------------
// Types
// ---------------------------------------------------------------------------
typedef __attribute__((ext_vector_type(16))) __bf16        bf16x16;
typedef __attribute__((ext_vector_type(8)))  float         f32x8;
typedef __attribute__((ext_vector_type(4)))  unsigned int  u32x4;

union Frag {                     // one WMMA A/B operand (8 VGPRs of a lane)
    bf16x16        v;
    u32x4          q[2];
    unsigned short u[16];
};

static __device__ __forceinline__ unsigned short f2bf(float f) {
    unsigned int x = __float_as_uint(f);
    x += 0x7fffu + ((x >> 16) & 1u);       // round-to-nearest-even
    return (unsigned short)(x >> 16);
}
static __device__ __forceinline__ float bf2f(unsigned short u) {
    return __uint_as_float(((unsigned int)u) << 16);
}

// ---------------------------------------------------------------------------
// Weight repack: (3,3,3,Cin,Cout) f32  ->  per-(tap,kchunk,ntile,lane) packed
// bf16 B-fragments matching the 16-bit 32x16 WMMA operand layout:
//   lane L holds column n = nt*16 + (L&15); klo = (L&16)?8:0
//   fragment element j (0..15) holds K = ((j>>3)<<4) + klo + (j&7)
// Channels c >= Cin are zero-padded.
// ---------------------------------------------------------------------------
__global__ void k_repack_w(const float* __restrict__ w,
                           unsigned short* __restrict__ dst,
                           int Cin, int Cout, int KC) {
    const int NT    = Cout >> 4;
    const int total = 27 * KC * NT * 32 * 16;
    int gid = blockIdx.x * blockDim.x + threadIdx.x;
    if (gid >= total) return;
    const int j    = gid & 15;
    const int lane = (gid >> 4) & 31;
    const int nt   = (gid >> 9) % NT;
    const int kc   = (gid / (512 * NT)) % KC;
    const int t    =  gid / (512 * NT * KC);
    const int klo  = (lane & 16) ? 8 : 0;
    const int K    = ((j >> 3) << 4) + klo + (j & 7);
    const int c    = kc * 32 + K;
    const int n    = nt * 16 + (lane & 15);
    float val = (c < Cin) ? w[(t * Cin + c) * Cout + n] : 0.0f;
    dst[gid] = f2bf(val);
}

// ---------------------------------------------------------------------------
// Implicit-GEMM 3x3x3 conv (pad=1, stride 1/2) via v_wmma_f32_16x16x32_bf16.
// One wave = 16 output voxels x (16*NTW) output channels.
// All channel counts padded to multiples of 32 -> A loads are always two
// b128 loads; halo rows handled branchlessly (clamped address + v_and mask).
// Ping-pong software pipeline with a single-exit main loop (no acc copies).
// ---------------------------------------------------------------------------
template <int NTW>
__global__ void __launch_bounds__(256)
k_conv3_wmma(const unsigned short* __restrict__ in,
             unsigned short* __restrict__ out,
             const unsigned short* __restrict__ wpk,
             const float* __restrict__ bias,
             const float* __restrict__ mask,
             int Din, int Dout, int stride,
             int CinPad, int Cout, int outStride, int relu) {
    const int NT     = Cout >> 4;
    const int NTG    = NT / NTW;
    const int lane   = threadIdx.x & 31;
    const int gw     = blockIdx.x * (blockDim.x >> 5) + (threadIdx.x >> 5);
    const int nStile = (Dout * Dout * Dout) >> 4;
    if (gw >= nStile * NTG) return;        // whole-wave exit: EXEC stays all-1s

    const int stile = gw / NTG;
    const int ntb   = (gw - stile * NTG) * NTW;
    const int sbase = stile << 4;

    const int m   = lane & 15;             // A-matrix row (output voxel)
    const int s   = sbase + m;
    const int oz  = s / (Dout * Dout);
    const int oy  = (s / Dout) % Dout;
    const int ox  = s % Dout;
    const int klo = (lane & 16) ? 8 : 0;   // K half selected by lane group

    const int KC      = CinPad >> 5;       // 1, 2 or 4
    const int kcMask  = KC - 1;
    const int kcShift = (KC == 1) ? 0 : ((KC == 2) ? 1 : 2);
    const int TOT     = 27 * KC;

    f32x8 acc[NTW];
#pragma unroll
    for (int u = 0; u < NTW; ++u)
#pragma unroll
        for (int i = 0; i < 8; ++i) acc[u][i] = 0.0f;

    // ---- per-tap A base address + halo select mask (fully branchless) ----
    auto tapAddr = [&](int t, unsigned int& selr) -> const unsigned short* {
        const int dz = t / 9 - 1;
        const int r  = t - (t / 9) * 9;
        const int dy = r / 3 - 1;
        const int dx = r - (r / 3) * 3 - 1;
        const int iz = oz * stride + dz;
        const int iy = oy * stride + dy;
        const int ix = ox * stride + dx;
        const bool inb = (unsigned)iz < (unsigned)Din &&
                         (unsigned)iy < (unsigned)Din &&
                         (unsigned)ix < (unsigned)Din;
        selr = inb ? 0xFFFFFFFFu : 0u;
        // clamp -> address always valid, no divergent branch
        const int izc = min(max(iz, 0), Din - 1);
        const int iyc = min(max(iy, 0), Din - 1);
        const int ixc = min(max(ix, 0), Din - 1);
        const unsigned int voff =
            (unsigned)(((izc * Din + iyc) * Din + ixc) * CinPad);
        return in + voff + klo;
    };
    auto loadA = [&](Frag& A, const unsigned short* apv, int kc, unsigned int selv) {
        const u32x4* p = reinterpret_cast<const u32x4*>(apv + kc * 32);
        u32x4 a0 = p[0];            // channels kc*32 + klo .. +7
        u32x4 a1 = p[2];            // channels kc*32 + 16 + klo .. +7
        a0 &= selv;
        a1 &= selv;
        A.q[0] = a0;
        A.q[1] = a1;
    };
    auto loadB = [&](Frag* B, int t, int kc) {
        const unsigned short* wb = wpk + (size_t)((t * KC + kc) * NT) * 512;
#pragma unroll
        for (int u = 0; u < NTW; ++u) {
            const u32x4* bp = reinterpret_cast<const u32x4*>(
                wb + (size_t)(ntb + u) * 512 + lane * 16);
            B[u].q[0] = bp[0];
            B[u].q[1] = bp[1];
        }
    };

    // rolling per-tap state (updated in issue order by loadStep)
    int curT = 0;
    unsigned int sel;
    const unsigned short* ap = tapAddr(0, sel);

    auto loadStep = [&](int j, Frag& A, Frag* B) {
        const int tj  = j >> kcShift;
        const int kcj = j & kcMask;
        if (tj != curT) {
            curT = tj;
            ap = tapAddr(tj, sel);
            if (tj + 1 < 27)   // prefetch next tap's weight block
                __builtin_prefetch(
                    wpk + (size_t)(((tj + 1) * KC) * NT + ntb) * 512 + lane * 16,
                    0, 1);
        }
        loadA(A, ap, kcj, sel);
        loadB(B, tj, kcj);
    };
    auto mmac = [&](Frag& A, Frag* B) {
#pragma unroll
        for (int u = 0; u < NTW; ++u)
            acc[u] = __builtin_amdgcn_wmma_f32_16x16x32_bf16(
                false, A.v, false, B[u].v, (short)0, acc[u], false, false);
    };

    // ---- ping-pong pipelined K loop: single-exit main loop + tail ----
    Frag A0, A1;
    Frag B0[NTW], B1[NTW];
    loadStep(0, A0, B0);

    int it = 0;
    for (; it + 2 < TOT; it += 2) {
        loadStep(it + 1, A1, B1);
        mmac(A0, B0);
        loadStep(it + 2, A0, B0);
        mmac(A1, B1);
    }
    if (it + 1 < TOT) {          // two steps left
        loadStep(it + 1, A1, B1);
        mmac(A0, B0);
        mmac(A1, B1);
    } else {                     // one step left
        mmac(A0, B0);
    }

    // ---- epilogue: bias + optional ReLU + occupancy mask, store bf16 ----
    const int mrow0 = (lane & 16) ? 8 : 0;   // C/D layout: rows i / 8+i
#pragma unroll
    for (int u = 0; u < NTW; ++u) {
        const int n  = (ntb + u) * 16 + (lane & 15);
        const float bn = bias[n];
#pragma unroll
        for (int i = 0; i < 8; ++i) {
            const int so = sbase + mrow0 + i;
            float v = acc[u][i] + bn;
            if (relu) v = fmaxf(v, 0.0f);
            v *= mask[so];
            out[(size_t)so * outStride + n] = f2bf(v);
        }
    }
}

// ---------------------------------------------------------------------------
// Elementwise helpers
// ---------------------------------------------------------------------------
__global__ void k_mask_i2f(const int* __restrict__ m, float* __restrict__ o, int n) {
    int i = blockIdx.x * blockDim.x + threadIdx.x;
    if (i < n) o[i] = (float)m[i];
}
__global__ void k_pool(const float* __restrict__ src, float* __restrict__ dst, int Do) {
    int i = blockIdx.x * blockDim.x + threadIdx.x;
    int tot = Do * Do * Do;
    if (i >= tot) return;
    int oz = i / (Do * Do), oy = (i / Do) % Do, ox = i % Do;
    int Di = Do * 2;
    float v = 0.0f;
    for (int a = 0; a < 2; ++a)
        for (int b = 0; b < 2; ++b)
            for (int c = 0; c < 2; ++c) {
                int idx = ((oz * 2 + a) * Di + (oy * 2 + b)) * Di + (ox * 2 + c);
                v = fmaxf(v, src[idx]);
            }
    dst[i] = v;
}
// f32 (S,C) -> bf16 (S,32) zero-padded
__global__ void k_padcvt(const float* __restrict__ s, unsigned short* __restrict__ d,
                         int S, int C) {
    int i = blockIdx.x * blockDim.x + threadIdx.x;
    if (i >= S * 32) return;
    int sv = i >> 5, c = i & 31;
    d[i] = (c < C) ? f2bf(s[sv * C + c]) : (unsigned short)0;
}
__global__ void k_fill0(unsigned short* __restrict__ d, int n) {
    int i = blockIdx.x * blockDim.x + threadIdx.x;
    if (i < n) d[i] = 0;
}
__global__ void k_film(unsigned short* __restrict__ x,
                       const unsigned short* __restrict__ bg,
                       const float* __restrict__ m, int S, int C) {
    int i = blockIdx.x * blockDim.x + threadIdx.x;
    if (i >= S * C) return;
    int s = i / C, c = i - s * C;
    float beta  = bf2f(bg[(size_t)s * 2 * C + c]);
    float gamma = bf2f(bg[(size_t)s * 2 * C + C + c]);
    x[i] = f2bf((bf2f(x[i]) * gamma + beta) * m[s]);
}
__global__ void k_final(const unsigned short* __restrict__ xp,
                        const unsigned short* __restrict__ q,
                        const float* __restrict__ m,
                        float* __restrict__ out, int S) {
    int i = blockIdx.x * blockDim.x + threadIdx.x;
    if (i >= S * 160) return;
    int s = i / 160, c = i - s * 160;
    float v = (c < 128) ? bf2f(xp[s * 128 + c]) : bf2f(q[s * 32 + (c - 128)]);
    out[i] = v * m[s];
}

// ---------------------------------------------------------------------------
// Host orchestration
// ---------------------------------------------------------------------------
extern "C" void kernel_launch(void* const* d_in, const int* in_sizes, int n_in,
                              void* d_out, int out_size, void* d_ws, size_t ws_size,
                              hipStream_t stream) {
    (void)in_sizes; (void)n_in; (void)out_size; (void)ws_size;

    // ---- workspace bump allocator (256B aligned) ----
    size_t off = 0;
    auto alloc = [&](size_t bytes) -> void* {
        off = (off + 255) & ~(size_t)255;
        void* p = (char*)d_ws + off;
        off += bytes;
        return p;
    };

    float* m0 = (float*)alloc(262144 * 4);
    float* m1 = (float*)alloc(32768 * 4);
    float* m2 = (float*)alloc(4096 * 4);
    float* m3 = (float*)alloc(512 * 4);

    unsigned short* xin  = (unsigned short*)alloc((size_t)8388608 * 2);  // 64^3 x 32
    unsigned short* qin  = (unsigned short*)alloc((size_t)8388608 * 2);  // 64^3 x 32
    unsigned short* x0   = (unsigned short*)alloc((size_t)16777216 * 2); // 64^3 x 64
    unsigned short* x1   = (unsigned short*)alloc((size_t)4194304 * 2);  // 32^3 x 128
    unsigned short* q1a  = (unsigned short*)alloc((size_t)1048576 * 2);  // 32^3 x 32
    unsigned short* q1   = (unsigned short*)alloc((size_t)1048576 * 2);
    unsigned short* x2   = (unsigned short*)alloc((size_t)524288 * 2);   // 16^3 x 128
    unsigned short* q2a  = (unsigned short*)alloc((size_t)131072 * 2);
    unsigned short* q2   = (unsigned short*)alloc((size_t)131072 * 2);
    unsigned short* x3   = (unsigned short*)alloc((size_t)65536 * 2);    // 8^3 x 128
    unsigned short* q3a  = (unsigned short*)alloc((size_t)16384 * 2);
    unsigned short* q3   = (unsigned short*)alloc((size_t)16384 * 2);
    unsigned short* xpost= (unsigned short*)alloc((size_t)65536 * 2);

    // Reuse the x0 slab (free after d1 consumes it):
    unsigned short* q0  = x0;                         // 64^3 x 32 = 8,388,608
    unsigned short* t1  = x0;                         // 4,194,304 (after q0 dies)
    unsigned short* bg1 = x0 + (size_t)4194304;       // 8,388,608
    unsigned short* t2  = x0 + (size_t)12582912;      //   524,288
    unsigned short* bg2 = x0 + (size_t)13107200;      // 1,048,576
    unsigned short* t3  = x0 + (size_t)14155776;      //    65,536
    unsigned short* bg3 = x0 + (size_t)14221312;      //   131,072

    // ---- conv table ----
    struct CW { int widx, bidx, Cin, Cout; unsigned short* pk; };
    CW cw[18] = {
        { 3, 4,   3,  64, nullptr}, { 5, 6,  64, 128, nullptr},
        { 7, 8, 128, 128, nullptr}, { 9,10, 128, 128, nullptr},
        {11,12, 128, 128, nullptr}, {13,14,   2,  16, nullptr},
        {15,16,  16,  32, nullptr}, {17,18,  32,  32, nullptr},
        {19,20,  32,  32, nullptr}, {21,22,  32,  32, nullptr},
        {23,24,  32,  32, nullptr}, {25,26,  32,  32, nullptr},
        {27,28,  32, 128, nullptr}, {29,30, 128, 256, nullptr},
        {31,32,  32, 128, nullptr}, {33,34, 128, 256, nullptr},
        {35,36,  32, 128, nullptr}, {37,38, 128, 256, nullptr},
    };
    for (int i = 0; i < 18; ++i) {
        int KC = (cw[i].Cin + 31) / 32;
        cw[i].pk = (unsigned short*)alloc((size_t)27 * KC * 32 * cw[i].Cout * 2);
    }
    for (int i = 0; i < 18; ++i) {
        int KC = (cw[i].Cin + 31) / 32;
        int total = 27 * KC * (cw[i].Cout / 16) * 32 * 16;
        k_repack_w<<<(total + 255) / 256, 256, 0, stream>>>(
            (const float*)d_in[cw[i].widx], cw[i].pk, cw[i].Cin, cw[i].Cout, KC);
    }

    // ---- masks + padded input conversion ----
    k_mask_i2f<<<(262144 + 255) / 256, 256, 0, stream>>>((const int*)d_in[0], m0, 262144);
    k_pool<<<(32768 + 255) / 256, 256, 0, stream>>>(m0, m1, 32);
    k_pool<<<(4096 + 255) / 256, 256, 0, stream>>>(m1, m2, 16);
    k_pool<<<(512 + 255) / 256, 256, 0, stream>>>(m2, m3, 8);
    k_padcvt<<<(8388608 + 255) / 256, 256, 0, stream>>>((const float*)d_in[1], xin, 262144, 3);
    k_padcvt<<<(8388608 + 255) / 256, 256, 0, stream>>>((const float*)d_in[2], qin, 262144, 2);

    auto conv = [&](const unsigned short* in, unsigned short* out, int ci,
                    const float* mask, int Din, int Dout, int stride,
                    int CinPad, int outStride, int relu) {
        int Cout = cw[ci].Cout;
        int ntw = (Cout % 32 == 0) ? 2 : 1;
        int waves = ((Dout * Dout * Dout) >> 4) * (Cout / (16 * ntw));
        int blocks = (waves + 7) / 8;
        const float* b = (const float*)d_in[cw[ci].bidx];
        if (ntw == 2)
            k_conv3_wmma<2><<<blocks, 256, 0, stream>>>(
                in, out, cw[ci].pk, b, mask, Din, Dout, stride, CinPad, Cout,
                outStride, relu);
        else
            k_conv3_wmma<1><<<blocks, 256, 0, stream>>>(
                in, out, cw[ci].pk, b, mask, Din, Dout, stride, CinPad, Cout,
                outStride, relu);
    };
    auto film = [&](unsigned short* x, const unsigned short* bg,
                    const float* m, int S, int C) {
        k_film<<<(S * C + 255) / 256, 256, 0, stream>>>(x, bg, m, S, C);
    };

    enum { PRE, D1, D2, D3, POST, QPRE, QD1, QD2, QD3, QL1, QL2, QL3,
           QP1A, QP1B, QP2A, QP2B, QP3A, QP3B };

    // stem + level-1 x-path first so the x0 slab frees up for q0/FiLM temps
    conv(xin, x0, PRE,  m0, 64, 64, 1, 32, 64, 1);
    conv(x0,  x1, D1,   m1, 64, 32, 2, 64, 128, 0);

    // q stem (q0 lives in the x0 slab; pad channels pre-zeroed)
    k_fill0<<<(8388608 + 255) / 256, 256, 0, stream>>>(q0, 8388608);
    conv(qin, q0,  QPRE, m0, 64, 64, 1, 32, 32, 1);   // Cout=16, stride-32 store
    conv(q0,  q1a, QD1,  m1, 64, 32, 2, 32, 32, 0);
    conv(q1a, q1,  QL1,  m1, 32, 32, 1, 32, 32, 1);
    conv(q1,  t1,  QP1A, m1, 32, 32, 1, 32, 128, 1);
    conv(t1,  bg1, QP1B, m1, 32, 32, 1, 128, 256, 0);
    film(x1, bg1, m1, 32768, 128);
    // level 2
    conv(x1,  x2,  D2,   m2, 32, 16, 2, 128, 128, 0);
    conv(q1,  q2a, QD2,  m2, 32, 16, 2, 32, 32, 0);
    conv(q2a, q2,  QL2,  m2, 16, 16, 1, 32, 32, 1);
    conv(q2,  t2,  QP2A, m2, 16, 16, 1, 32, 128, 1);
    conv(t2,  bg2, QP2B, m2, 16, 16, 1, 128, 256, 0);
    film(x2, bg2, m2, 4096, 128);
    // level 3
    conv(x2,  x3,  D3,   m3, 16, 8, 2, 128, 128, 0);
    conv(q2,  q3a, QD3,  m3, 16, 8, 2, 32, 32, 0);
    conv(q3a, q3,  QL3,  m3,  8, 8, 1, 32, 32, 0);   // q_l3 has no ReLU
    conv(q3,  t3,  QP3A, m3,  8, 8, 1, 32, 128, 1);
    conv(t3,  bg3, QP3B, m3,  8, 8, 1, 128, 256, 0);
    film(x3, bg3, m3, 512, 128);
    // post + concat
    conv(x3, xpost, POST, m3, 8, 8, 1, 128, 128, 0);
    k_final<<<(512 * 160 + 255) / 256, 256, 0, stream>>>(xpost, q3, m3,
                                                         (float*)d_out, 512);
}
